// LieLayer_90958817394732
// MI455X (gfx1250) — compile-verified, matrix operand order checked by reading, not run on hardware
//
#include <hip/hip_runtime.h>
#include <hip/hip_bf16.h>

typedef __bf16 bf16;
typedef __attribute__((ext_vector_type(16))) __bf16 v16bf;
typedef __attribute__((ext_vector_type(4)))  __bf16 v4bf;
typedef __attribute__((ext_vector_type(8)))  float  v8f;

#define B_DIM 4096
#define N_DIM 1024
#define P_DIM 32
#define E_DIM 16
#define C_DIM (N_DIM * E_DIM)   // 16384 fused output columns (i*E + e)

// ---------------------------------------------------------------------------
// Kernel A: x [B, N] f32 -> bf16
// ---------------------------------------------------------------------------
__global__ void cvt_x_bf16(const float* __restrict__ x, bf16* __restrict__ xb) {
    long idx = ((long)blockIdx.x * blockDim.x + threadIdx.x) * 4;
    float4 v = *(const float4*)(x + idx);
    v4bf o;
    o.x = (bf16)v.x; o.y = (bf16)v.y; o.z = (bf16)v.z; o.w = (bf16)v.w;
    *(v4bf*)(xb + idx) = o;
}

// ---------------------------------------------------------------------------
// Kernel B: W[(i*E + e), j] = sum_p params[p,e] * T[p, i*N + j], stored bf16.
// One thread per q = i*N + j; params broadcast via LDS.
// ---------------------------------------------------------------------------
__global__ void build_w_bf16(const float* __restrict__ params,
                             const float* __restrict__ T,
                             bf16* __restrict__ W) {
    __shared__ float sp[P_DIM * E_DIM];          // 512 floats
    int tid = threadIdx.x;                        // blockDim.x == 512
    sp[tid] = params[tid];
    __syncthreads();

    long q = (long)blockIdx.x * 512 + tid;        // 0 .. N*N-1
    int i = (int)(q >> 10);
    int j = (int)(q & (N_DIM - 1));

    float acc[E_DIM];
#pragma unroll
    for (int e = 0; e < E_DIM; ++e) acc[e] = 0.f;

#pragma unroll
    for (int p = 0; p < P_DIM; ++p) {
        float t = T[(long)p * N_DIM * N_DIM + q];
#pragma unroll
        for (int e = 0; e < E_DIM; ++e) acc[e] += sp[p * E_DIM + e] * t;
    }

    long base = (long)i * C_DIM + j;              // + e*N_DIM selects column e
#pragma unroll
    for (int e = 0; e < E_DIM; ++e)
        W[base + (long)e * N_DIM] = (bf16)acc[e];
}

// ---------------------------------------------------------------------------
// Kernel C: out[4096, 16384] = X[4096,1024] @ W[16384,1024]^T, swish epilogue.
// 8 waves/block; each wave computes a 64x64 tile via 4x4 WMMA accumulators.
// A-frag as loaded here (16x32; contiguous k, not the ISA table order): lane&15 = row M, lane>>4 picks K half,
// 16 contiguous bf16 per lane. B-frag identical by symmetry (both row-major,
// GEMM is X @ W^T).
// ---------------------------------------------------------------------------
__global__ void __launch_bounds__(256)
gemm_swish(const bf16* __restrict__ X,   // [B_DIM, N_DIM]
           const bf16* __restrict__ W,   // [C_DIM, N_DIM]
           float* __restrict__ out) {    // [B_DIM, C_DIM]
    const int lane = threadIdx.x & 31;
    const int wid  = threadIdx.x >> 5;
    const int wm   = wid >> 2;            // 0..1
    const int wn   = wid & 3;             // 0..3
    const int l15  = lane & 15;
    const int khalf = (lane >> 4) * 16;   // 0 or 16

    const int row0 = blockIdx.y * 128 + wm * 64;   // output rows (b)
    const int col0 = blockIdx.x * 256 + wn * 64;   // output cols (i*E+e)

    const bf16* xp = X + (long)(row0 + l15) * N_DIM + khalf;
    const bf16* wp = W + (long)(col0 + l15) * N_DIM + khalf;

    v8f acc[4][4] = {};

    for (int k0 = 0; k0 < N_DIM; k0 += 32) {
        v16bf a[4], b[4];
#pragma unroll
        for (int mi = 0; mi < 4; ++mi)
            a[mi] = *(const v16bf*)(xp + (long)mi * 16 * N_DIM + k0);
#pragma unroll
        for (int ni = 0; ni < 4; ++ni)
            b[ni] = *(const v16bf*)(wp + (long)ni * 16 * N_DIM + k0);
#pragma unroll
        for (int mi = 0; mi < 4; ++mi)
#pragma unroll
            for (int ni = 0; ni < 4; ++ni)
                acc[mi][ni] = __builtin_amdgcn_wmma_f32_16x16x32_bf16(
                    false, a[mi], false, b[ni],
                    (short)0, acc[mi][ni], false, false);
    }

    // D layout: VGPR r -> row (r) for lanes 0-15, row (8+r) for lanes 16-31;
    // column = lane&15 within the 16x16 tile.
    const int mrow_off = (lane >> 4) * 8;
#pragma unroll
    for (int mi = 0; mi < 4; ++mi) {
#pragma unroll
        for (int ni = 0; ni < 4; ++ni) {
#pragma unroll
            for (int r = 0; r < 8; ++r) {
                int grow = row0 + mi * 16 + mrow_off + r;
                int gcol = col0 + ni * 16 + l15;
                float v = acc[mi][ni][r];
                // swish = v * sigmoid(v); v_exp_f32 + v_rcp_f32 + v_mul
                // (avoid the IEEE v_div_scale/div_fmas sequence)
                float s = v * __builtin_amdgcn_rcpf(1.0f + __expf(-v));
                out[(long)grow * C_DIM + gcol] = s;
            }
        }
    }
}

// ---------------------------------------------------------------------------
extern "C" void kernel_launch(void* const* d_in, const int* in_sizes, int n_in,
                              void* d_out, int out_size, void* d_ws, size_t ws_size,
                              hipStream_t stream) {
    const float* x      = (const float*)d_in[0];   // [4096, 1024]
    const float* params = (const float*)d_in[1];   // [32, 16]
    const float* T      = (const float*)d_in[2];   // [32, 1024, 1024]
    float* out          = (float*)d_out;           // [4096, 1024, 16]

    bf16* xb = (bf16*)d_ws;                                             // 8 MB
    bf16* Wb = (bf16*)((char*)d_ws + (size_t)B_DIM * N_DIM * sizeof(bf16)); // 32 MB

    // A: x -> bf16
    cvt_x_bf16<<<(B_DIM * N_DIM / 4) / 256, 256, 0, stream>>>(x, xb);
    // B: build fused filter matrix in bf16
    build_w_bf16<<<(N_DIM * N_DIM) / 512, 512, 0, stream>>>(params, T, Wb);
    // C: WMMA GEMM + swish
    dim3 grid(C_DIM / 256, B_DIM / 128);
    gemm_swish<<<grid, 256, 0, stream>>>(xb, Wb, out);
}
